// latent_rot_decode_91070486544796
// MI455X (gfx1250) — matrix operand order repaired and checked by reading, not compiled, not run on hardware
//
#include <hip/hip_runtime.h>

typedef _Float16 v16h __attribute__((ext_vector_type(16)));
typedef _Float16 v8h  __attribute__((ext_vector_type(8)));
typedef float    v8f  __attribute__((ext_vector_type(8)));
typedef float    v4f  __attribute__((ext_vector_type(4)));

#define LATENT 64
#define HID    40
#define OMEGA_D 9
#define TRANZ_D 3
#define KCOLS  262144
#define NTILES (KCOLS/16)
#define WROWS  112          // per-branch padded weight rows: 48 (L1) + 48 (L2) + 16 (L3)
#define SR     20           // LDS scratch row stride in floats (80B: 16B aligned, bank-spread)

__device__ __forceinline__ void mat3mul(float* C, const float* A, const float* B) {
#pragma unroll
  for (int r = 0; r < 3; ++r)
#pragma unroll
    for (int c = 0; c < 3; ++c)
      C[r*3+c] = fmaf(A[r*3+0], B[0*3+c], fmaf(A[r*3+1], B[1*3+c], A[r*3+2]*B[2*3+c]));
}

// A-operand fragment (16x32 f16): lane<16 -> M=lane, K = {0..7,16..23}+32*chunk
//                                 lane>=16 -> M=lane-16, K = {8..15,24..31}+32*chunk
__device__ __forceinline__ v16h load_afrag(const _Float16* wmat, int mtile, int chunk, int lane) {
  const int m  = mtile*16 + (lane & 15);
  const int kb = chunk*32 + ((lane & 16) ? 8 : 0);
  const _Float16* p = wmat + m*64 + kb;
  v8h lo = *(const v8h*)(p);
  v8h hi = *(const v8h*)(p + 16);
  v16h a;
#pragma unroll
  for (int e = 0; e < 8; ++e) { a[e] = lo[e]; a[8+e] = hi[e]; }
  return a;
}

// B-operand fragment: this lane holds column N=lane%16; K = {0..7,16..23} (lane<16) or {8..15,24..31} (lane>=16), f32 -> f16
__device__ __forceinline__ v16h cvt_bfrag_lds(const float* p, int stride, int lane) {
  const int kb = (lane & 16) ? 8 : 0;
  v16h b;
#pragma unroll
  for (int e = 0; e < 8; ++e) {
    b[e]   = (_Float16)p[(kb + e)*stride];
    b[8+e] = (_Float16)p[(kb + 16 + e)*stride];
  }
  return b;
}

__device__ __forceinline__ v16h cvt_bfrag_global_nt(const float* p, int stride, int lane) {
  const int kb = (lane & 16) ? 8 : 0;
  v16h b;
#pragma unroll
  for (int e = 0; e < 8; ++e) {
    b[e]   = (_Float16)__builtin_nontemporal_load(p + (kb + e)*stride);
    b[8+e] = (_Float16)__builtin_nontemporal_load(p + (kb + 16 + e)*stride);
  }
  return b;
}

__global__ __launch_bounds__(256) void latent_rot_decode_kernel(
    const float* __restrict__ x,
    const float* __restrict__ Wo1, const float* __restrict__ bo1,
    const float* __restrict__ Wo2, const float* __restrict__ bo2,
    const float* __restrict__ Wo3, const float* __restrict__ bo3,
    const float* __restrict__ Wt1, const float* __restrict__ bt1,
    const float* __restrict__ Wt2, const float* __restrict__ bt2,
    const float* __restrict__ Wt3, const float* __restrict__ bt3,
    float* __restrict__ out)
{
  __shared__ __align__(16) _Float16 sW[2][WROWS][64];  // padded f16 weights
  __shared__ float                  sB[2][WROWS];      // padded f32 biases
  __shared__ __align__(16) float    sS[8][48*SR];      // per-wave transpose scratch

  const int tid  = threadIdx.x;
  const int lane = tid & 31;
  const int wave = tid >> 5;

  // ---- cooperative weight pad + f32->f16 convert into LDS ----
  for (int i = tid; i < 2*WROWS*64; i += 256) {
    int br  = i / (WROWS*64);
    int rem = i - br*(WROWS*64);
    int row = rem >> 6, k = rem & 63;
    float v = 0.0f;
    if (row < 48) {                       // layer1: HIDxLATENT real
      if (row < HID) v = (br ? Wt1 : Wo1)[row*LATENT + k];
    } else if (row < 96) {                // layer2: HIDxHID real
      int r = row - 48;
      if (r < HID && k < HID) v = (br ? Wt2 : Wo2)[r*HID + k];
    } else {                              // layer3: {9,3}xHID real
      int r = row - 96;
      int nrow = br ? TRANZ_D : OMEGA_D;
      if (r < nrow && k < HID) v = (br ? Wt3 : Wo3)[r*HID + k];
    }
    sW[br][row][k] = (_Float16)v;
  }
  for (int i = tid; i < 2*WROWS; i += 256) {
    int br = i / WROWS, row = i - br*WROWS;
    float v = 0.0f;
    if (row < 48)      { if (row < HID) v = (br ? bt1 : bo1)[row]; }
    else if (row < 96) { int r = row-48; if (r < HID) v = (br ? bt2 : bo2)[r]; }
    else               { int r = row-96; int nrow = br ? TRANZ_D : OMEGA_D;
                         if (r < nrow) v = (br ? bt3 : bo3)[r]; }
    sB[br][row] = v;
  }
  __syncthreads();

  float* scr = sS[wave];
  const int cb = lane & 15;             // this lane's column within the tile (C layout)
  const int ro = (lane & 16) ? 8 : 0;   // C-layout row offset for upper lane half

  for (int tile = blockIdx.x*8 + wave; tile < NTILES; tile += gridDim.x*8) {
    const int col0 = tile * 16;

    // x tile -> B fragments (two K-chunks of 32 feature rows; lane%16 = N column)
    v16h bx0 = cvt_bfrag_global_nt(x + col0 + cb, KCOLS, lane);
    v16h bx1 = cvt_bfrag_global_nt(x + 32*KCOLS + col0 + cb, KCOLS, lane);

#pragma unroll
    for (int br = 0; br < 2; ++br) {
      const _Float16* W  = &sW[br][0][0];
      const float*    Bv = &sB[br][0];

      // ---- layer 1: (48x64) @ (64x16) ----
#pragma unroll
      for (int mt = 0; mt < 3; ++mt) {
        v8f acc;
#pragma unroll
        for (int j = 0; j < 8; ++j) acc[j] = Bv[mt*16 + j + ro];
        acc = __builtin_amdgcn_wmma_f32_16x16x32_f16(false, load_afrag(W, mt, 0, lane),
                                                     false, bx0, (short)0, acc, false, false);
        acc = __builtin_amdgcn_wmma_f32_16x16x32_f16(false, load_afrag(W, mt, 1, lane),
                                                     false, bx1, (short)0, acc, false, false);
#pragma unroll
        for (int j = 0; j < 8; ++j) scr[(mt*16 + j + ro)*SR + cb] = fmaxf(acc[j], 0.0f);
      }

      // h -> B fragments (rows 40..47 are exact zeros from padding; rows 48..63 forced 0)
      v16h b20 = cvt_bfrag_lds(scr + cb, SR, lane);
      v16h b21;
      {
        const float* q = scr + (32 + ((lane & 16) ? 8 : 0))*SR + cb;
#pragma unroll
        for (int e = 0; e < 8; ++e) { b21[e] = (_Float16)q[e*SR]; b21[8+e] = (_Float16)0.0f; }
      }

      // ---- layer 2: (48x64pad) @ (64x16) ----
#pragma unroll
      for (int mt = 0; mt < 3; ++mt) {
        v8f acc;
#pragma unroll
        for (int j = 0; j < 8; ++j) acc[j] = Bv[48 + mt*16 + j + ro];
        acc = __builtin_amdgcn_wmma_f32_16x16x32_f16(false, load_afrag(W + 48*64, mt, 0, lane),
                                                     false, b20, (short)0, acc, false, false);
        acc = __builtin_amdgcn_wmma_f32_16x16x32_f16(false, load_afrag(W + 48*64, mt, 1, lane),
                                                     false, b21, (short)0, acc, false, false);
#pragma unroll
        for (int j = 0; j < 8; ++j) scr[(mt*16 + j + ro)*SR + cb] = fmaxf(acc[j], 0.0f);
      }

      v16h b30 = cvt_bfrag_lds(scr + cb, SR, lane);
      v16h b31;
      {
        const float* q = scr + (32 + ((lane & 16) ? 8 : 0))*SR + cb;
#pragma unroll
        for (int e = 0; e < 8; ++e) { b31[e] = (_Float16)q[e*SR]; b31[8+e] = (_Float16)0.0f; }
      }

      // ---- layer 3: (16x64pad) @ (64x16), single M tile ----
      v8f acc3;
#pragma unroll
      for (int j = 0; j < 8; ++j) acc3[j] = Bv[96 + j + ro];
      acc3 = __builtin_amdgcn_wmma_f32_16x16x32_f16(false, load_afrag(W + 96*64, 0, 0, lane),
                                                    false, b30, (short)0, acc3, false, false);
      acc3 = __builtin_amdgcn_wmma_f32_16x16x32_f16(false, load_afrag(W + 96*64, 0, 1, lane),
                                                    false, b31, (short)0, acc3, false, false);
#pragma unroll
      for (int j = 0; j < 8; ++j) acc3[j] = fmaxf(acc3[j], 0.0f);

      const int col = col0 + cb;
      if (br == 0) {
        // omega: rows 0..8 (relu'd), streamed out
        if (!(lane & 16)) {
#pragma unroll
          for (int j = 0; j < 8; ++j)
            __builtin_nontemporal_store(acc3[j], out + j*KCOLS + col);
        } else {
          __builtin_nontemporal_store(acc3[0], out + 8*KCOLS + col);
        }
        // stash full 16x16 (rows 9..15 are zeros) for the per-column expm
#pragma unroll
        for (int j = 0; j < 8; ++j) scr[(j + ro)*SR + cb] = acc3[j];

        if (!(lane & 16)) {
          float A[9];
#pragma unroll
          for (int i = 0; i < 9; ++i) A[i] = scr[i*SR + cb];
          // expm: scale by 2^-5, order-9 Taylor (Horner), square 5x
          float Bm[9];
#pragma unroll
          for (int i = 0; i < 9; ++i) Bm[i] = A[i] * 0.03125f;
          float T[9] = {1.f,0.f,0.f, 0.f,1.f,0.f, 0.f,0.f,1.f};
#pragma unroll
          for (int kk = 9; kk >= 1; --kk) {
            float P[9]; mat3mul(P, Bm, T);
            const float inv = 1.0f / (float)kk;
#pragma unroll
            for (int i = 0; i < 9; ++i) T[i] = P[i]*inv;
            T[0] += 1.0f; T[4] += 1.0f; T[8] += 1.0f;
          }
#pragma unroll
          for (int q = 0; q < 5; ++q) {
            float P[9]; mat3mul(P, T, T);
#pragma unroll
            for (int i = 0; i < 9; ++i) T[i] = P[i];
          }
#pragma unroll
          for (int i = 0; i < 9; ++i)
            __builtin_nontemporal_store(T[i], out + (9 + i)*KCOLS + col);
        }
      } else {
        // translation: rows 0..2
        if (!(lane & 16)) {
#pragma unroll
          for (int j = 0; j < 3; ++j)
            __builtin_nontemporal_store(acc3[j], out + (18 + j)*KCOLS + col);
        }
      }
    }
  }
}

extern "C" void kernel_launch(void* const* d_in, const int* in_sizes, int n_in,
                              void* d_out, int out_size, void* d_ws, size_t ws_size,
                              hipStream_t stream) {
  const float* x   = (const float*)d_in[0];
  const float* Wo1 = (const float*)d_in[1];  const float* bo1 = (const float*)d_in[2];
  const float* Wo2 = (const float*)d_in[3];  const float* bo2 = (const float*)d_in[4];
  const float* Wo3 = (const float*)d_in[5];  const float* bo3 = (const float*)d_in[6];
  const float* Wt1 = (const float*)d_in[7];  const float* bt1 = (const float*)d_in[8];
  const float* Wt2 = (const float*)d_in[9];  const float* bt2 = (const float*)d_in[10];
  const float* Wt3 = (const float*)d_in[11]; const float* bt3 = (const float*)d_in[12];
  float* out = (float*)d_out;

  dim3 grid(NTILES / 8);   // 2048 blocks x 8 waves = one wave32 per 16-column tile
  dim3 block(256);
  hipLaunchKernelGGL(latent_rot_decode_kernel, grid, block, 0, stream,
                     x, Wo1, bo1, Wo2, bo2, Wo3, bo3,
                     Wt1, bt1, Wt2, bt2, Wt3, bt3, out);
}
